// TPUOptimizedSSM_27599459844968
// MI455X (gfx1250) — matrix operand order repaired and checked by reading, not compiled, not run on hardware
//
#include <hip/hip_runtime.h>
#include <hip/hip_bf16.h>

// ---------------- problem constants ----------------
#define BATCH 8
#define SEQn  4096
#define DIMn  2048
#define HIDn  2048
#define BS    (BATCH * SEQn)     // 32768 rows
#define NSEG  32
#define SEGL  (SEQn / NSEG)      // 128 steps per segment

// CDNA5 async global->LDS copies (ASYNCcnt-tracked). Set to 0 to fall back to
// the synchronous VGPR-staged path.
#define USE_ASYNC_LDS 1

typedef __attribute__((ext_vector_type(16))) _Float16 v16h;
typedef __attribute__((ext_vector_type(8)))  float    v8f;
typedef __attribute__((ext_vector_type(4)))  unsigned int v4u;
typedef _Float16 h16;

// ---------------- async global->LDS helpers ----------------
__device__ __forceinline__ void asyncLoadB128(const void* g, void* ldsDst) {
#if USE_ASYNC_LDS
  // VDST VGPR carries the wave-relative LDS byte offset (low 32 bits of the
  // generic shared-aperture address, ISA 10.2); VADDR is the 64-bit global addr.
  unsigned ldsOff = (unsigned)(uintptr_t)ldsDst;
  asm volatile("global_load_async_to_lds_b128 %0, %1, off"
               :: "v"(ldsOff), "v"(g) : "memory");
#else
  *(v4u*)ldsDst = *(const v4u*)g;
#endif
}

__device__ __forceinline__ void asyncWait() {
#if USE_ASYNC_LDS
  asm volatile("s_wait_asynccnt 0" ::: "memory");
#endif
}

// ---------------- block reduction helper (wave32) ----------------
__device__ __forceinline__ float blockSum(float v, float* smem) {
#pragma unroll
  for (int o = 16; o > 0; o >>= 1) v += __shfl_xor(v, o, 32);
  const int wid = threadIdx.x >> 5;
  if ((threadIdx.x & 31) == 0) smem[wid] = v;
  __syncthreads();
  if (threadIdx.x == 0) {
    float t = 0.f;
#pragma unroll
    for (int i = 0; i < 8; ++i) t += smem[i];
    smem[8] = t;
  }
  __syncthreads();
  float r = smem[8];
  __syncthreads();
  return r;
}

// ---------------- fused LayerNorm + delta = exp(xn @ dt_kernel + dt_bias) ----
__global__ __launch_bounds__(256)
void ln_delta_kernel(const float* __restrict__ x, const float* __restrict__ gamma,
                     const float* __restrict__ beta, const float* __restrict__ dtk,
                     const float* __restrict__ dtb, h16* __restrict__ xn16,
                     float* __restrict__ delta) {
  __shared__ float red[9];
  const int row = blockIdx.x;                  // b*SEQ + t
  const float* xr = x + (size_t)row * DIMn;
  float v[8];
  float s = 0.f, ss = 0.f;
#pragma unroll
  for (int i = 0; i < 8; ++i) {
    v[i] = xr[threadIdx.x + i * 256];
    s += v[i];
    ss += v[i] * v[i];
  }
  const float mu  = blockSum(s, red) * (1.f / DIMn);
  const float var = blockSum(ss, red) * (1.f / DIMn) - mu * mu;
  const float rs  = rsqrtf(var + 1e-6f);
  h16* xo = xn16 + (size_t)row * DIMn;
  float dot = 0.f;
#pragma unroll
  for (int i = 0; i < 8; ++i) {
    const int d  = threadIdx.x + i * 256;
    const float xv = (v[i] - mu) * rs * gamma[d] + beta[d];
    xo[d] = (h16)xv;
    dot += xv * dtk[d];
  }
  const float dt = blockSum(dot, red);
  if (threadIdx.x == 0) delta[row] = __expf(dt + dtb[0]);
}

// ---------------- weight fp32 (K,N) -> f16 transposed (N,K) -----------------
__global__ __launch_bounds__(256)
void convertT_kernel(const float* __restrict__ W, h16* __restrict__ Wt) {
  __shared__ float tile[32][33];
  const int k0 = blockIdx.x * 32;
  const int n0 = blockIdx.y * 32;
  const int tx = threadIdx.x & 31;
  const int ty = threadIdx.x >> 5;  // 0..7
#pragma unroll
  for (int i = 0; i < 4; ++i)
    tile[ty + i * 8][tx] = W[(size_t)(k0 + ty + i * 8) * DIMn + n0 + tx];
  __syncthreads();
#pragma unroll
  for (int i = 0; i < 4; ++i)
    Wt[(size_t)(n0 + ty + i * 8) * DIMn + k0 + tx] = (h16)tile[tx][ty + i * 8];
}

// ---------------- WMMA GEMM: out[m][n] = sum_k A16[m][k] * Wt16[n][k] --------
// Block 256 thr (8 waves), tile 128x128, BK=32, double-buffered LDS with async
// global->LDS tile copies. Wave(wy 0..3, wx 0..1) owns a 32x64 output tile.
#define LDT 40   // padded f16 stride for LDS tiles (bank spread)

template <int MODE>
__global__ __launch_bounds__(256)
void wmma_gemm_kernel(const h16* __restrict__ A16, const h16* __restrict__ Wt16,
                      float* __restrict__ outF, h16* __restrict__ outH,
                      const float* __restrict__ bias) {
  __shared__ __align__(16) h16 As[2][128 * LDT];
  __shared__ __align__(16) h16 Bs[2][128 * LDT];
  const int tid  = threadIdx.x;
  const int lane = tid & 31;
  const int wave = tid >> 5;
  const int wy = wave >> 1, wx = wave & 1;
  const int mBase = blockIdx.y * 128;
  const int nBase = blockIdx.x * 128;
  const int m  = lane & 15;
  const int kh = lane >> 4;

  // per-thread tile-copy assignment: 512 chunks of 8 f16; thread t takes
  // chunks {t, t+256}; 4 consecutive threads cover one 64B row segment.
  int cr[2], cko[2];
#pragma unroll
  for (int i = 0; i < 2; ++i) {
    const int c = tid + i * 256;
    cr[i]  = c >> 2;
    cko[i] = (c & 3) * 8;
  }

  v8f acc[2][4];
#pragma unroll
  for (int i = 0; i < 2; ++i)
#pragma unroll
    for (int j = 0; j < 4; ++j)
#pragma unroll
      for (int e = 0; e < 8; ++e) acc[i][j][e] = 0.f;

  auto issueTile = [&](int kt, int buf) {
#pragma unroll
    for (int i = 0; i < 2; ++i) {
      asyncLoadB128(A16  + (size_t)(mBase + cr[i]) * 2048 + kt + cko[i],
                    &As[buf][cr[i] * LDT + cko[i]]);
      asyncLoadB128(Wt16 + (size_t)(nBase + cr[i]) * 2048 + kt + cko[i],
                    &Bs[buf][cr[i] * LDT + cko[i]]);
    }
  };

  issueTile(0, 0);
  asyncWait();
  __syncthreads();

  for (int it = 0; it < 64; ++it) {
    const int cur = it & 1;
    if (it < 63) issueTile((it + 1) * 32, cur ^ 1);  // overlap with compute

    // A fragment (16x32 f16): lane m rows; halves j0..7 -> K=kh*8+j,
    // j8..15 -> K=16+kh*8+j (ISA 7.12.2 layout).
    v16h af[2], bf[4];
#pragma unroll
    for (int mi = 0; mi < 2; ++mi) {
      const int row = wy * 32 + mi * 16 + m;
      ((v4u*)&af[mi])[0] = *(const v4u*)(&As[cur][row * LDT + kh * 8]);
      ((v4u*)&af[mi])[1] = *(const v4u*)(&As[cur][row * LDT + 16 + kh * 8]);
    }
    // B fragment (32x16 f16): lane col n; j0..7 -> K=kh*8+j, j8..15 -> K=16+kh*8+j.
#pragma unroll
    for (int ni = 0; ni < 4; ++ni) {
      const int rowB = wx * 64 + ni * 16 + m;
      ((v4u*)&bf[ni])[0] = *(const v4u*)(&Bs[cur][rowB * LDT + kh * 8]);
      ((v4u*)&bf[ni])[1] = *(const v4u*)(&Bs[cur][rowB * LDT + 16 + kh * 8]);
    }
#pragma unroll
    for (int mi = 0; mi < 2; ++mi)
#pragma unroll
      for (int ni = 0; ni < 4; ++ni)
        acc[mi][ni] = __builtin_amdgcn_wmma_f32_16x16x32_f16(
            false, af[mi], false, bf[ni], (short)0, acc[mi][ni], false, false);

    if (it < 63) {
      asyncWait();       // next tile's async LDS writes landed
      __syncthreads();   // visible block-wide; everyone done reading buf[cur]
    }
  }

  // Epilogue. C/D layout: VGPR r -> M = r (lanes 0-15) / r+8 (lanes 16-31).
  const int rowOff = mBase + wy * 32 + (lane >> 4) * 8;
  const int colOff = nBase + wx * 64 + (lane & 15);
#pragma unroll
  for (int mi = 0; mi < 2; ++mi)
#pragma unroll
    for (int ni = 0; ni < 4; ++ni) {
      const int col = colOff + ni * 16;
      const float bv = (MODE == 2) ? bias[col] : 0.f;
#pragma unroll
      for (int r = 0; r < 8; ++r) {
        const size_t idx = (size_t)(rowOff + mi * 16 + r) * 2048 + col;
        const float v = acc[mi][ni][r];
        if (MODE == 0) {                 // f16 output only (U)
          outH[idx] = (h16)v;
        } else if (MODE == 1) {          // f32 to d_out + f16 copy (outs)
          outF[idx] = v;
          outH[idx] = (h16)v;
        } else {                         // gate epilogue, in place on d_out
          const float g = v + bv;
          const float o = outF[idx];
          outF[idx] = o * (1.f / (1.f + __expf(-g)));
        }
      }
    }
}

// ---------------- segmented scan: s_t = s_{t-1}*exp(delta_t*A_h) + U_t -------
__global__ __launch_bounds__(256)
void scan_seg_kernel(const h16* __restrict__ U16, const float* __restrict__ delta,
                     const float* __restrict__ Av, float* __restrict__ segA,
                     float* __restrict__ segS) {
  __shared__ float dl[SEGL];
  const int h = blockIdx.x * 256 + threadIdx.x;
  const int b = blockIdx.y;
  const int seg = blockIdx.z;
  if (threadIdx.x < SEGL) dl[threadIdx.x] = delta[b * SEQn + seg * SEGL + threadIdx.x];
  __syncthreads();
  const float Ah = Av[h];
  const h16* up = U16 + ((size_t)b * SEQn + seg * SEGL) * HIDn + h;
  float aProd = 1.f, st = 0.f;
  for (int t = 0; t < SEGL; ++t) {
    const float dec = __expf(dl[t] * Ah);
    st = st * dec + (float)up[(size_t)t * HIDn];
    aProd *= dec;
  }
  const size_t o = ((size_t)seg * BATCH + b) * HIDn + h;
  segA[o] = aProd;
  segS[o] = st;
}

__global__ __launch_bounds__(256)
void scan_combine_kernel(const float* __restrict__ segA, const float* __restrict__ segS,
                         const float* __restrict__ context, float* __restrict__ segStart,
                         float* __restrict__ finalState) {
  const int h = blockIdx.x * 256 + threadIdx.x;
  const int b = blockIdx.y;
  float st = context[b * HIDn + h];
  for (int seg = 0; seg < NSEG; ++seg) {
    const size_t o = ((size_t)seg * BATCH + b) * HIDn + h;
    segStart[o] = st;
    st = st * segA[o] + segS[o];
  }
  finalState[b * HIDn + h] = st;
}

__global__ __launch_bounds__(256)
void scan_emit_kernel(const h16* __restrict__ U16, const float* __restrict__ delta,
                      const float* __restrict__ Av, const float* __restrict__ segStart,
                      h16* __restrict__ s16) {
  __shared__ float dl[SEGL];
  const int h = blockIdx.x * 256 + threadIdx.x;
  const int b = blockIdx.y;
  const int seg = blockIdx.z;
  if (threadIdx.x < SEGL) dl[threadIdx.x] = delta[b * SEQn + seg * SEGL + threadIdx.x];
  __syncthreads();
  const float Ah = Av[h];
  const size_t base = ((size_t)b * SEQn + seg * SEGL) * HIDn + h;
  const h16* up = U16 + base;
  h16* sp = s16 + base;
  float st = segStart[((size_t)seg * BATCH + b) * HIDn + h];
  for (int t = 0; t < SEGL; ++t) {
    const float dec = __expf(dl[t] * Ah);
    st = st * dec + (float)up[(size_t)t * HIDn];
    sp[(size_t)t * HIDn] = (h16)st;
  }
}

// ---------------- launch -----------------------------------------------------
extern "C" void kernel_launch(void* const* d_in, const int* in_sizes, int n_in,
                              void* d_out, int out_size, void* d_ws, size_t ws_size,
                              hipStream_t stream) {
  (void)in_sizes; (void)n_in; (void)out_size; (void)ws_size;
  const float* x   = (const float*)d_in[0];
  const float* ctx = (const float*)d_in[1];
  const float* Av  = (const float*)d_in[2];
  const float* Bw  = (const float*)d_in[3];
  const float* Cw  = (const float*)d_in[4];
  const float* dtk = (const float*)d_in[5];
  const float* dtb = (const float*)d_in[6];
  const float* Gw  = (const float*)d_in[7];
  const float* gb  = (const float*)d_in[8];
  const float* lns = (const float*)d_in[9];
  const float* lnb = (const float*)d_in[10];
  float* out = (float*)d_out;

  char* ws = (char*)d_ws;
  size_t off = 0;
  auto alloc = [&](size_t bytes) -> void* {
    void* p = ws + off;
    off = (off + bytes + 255) & ~(size_t)255;
    return p;
  };
  h16*   xn16  = (h16*)alloc((size_t)BS * DIMn * 2);
  h16*   U16   = (h16*)alloc((size_t)BS * HIDn * 2);
  h16*   s16   = (h16*)alloc((size_t)BS * HIDn * 2);
  h16*   o16   = (h16*)alloc((size_t)BS * DIMn * 2);
  h16*   Bt    = (h16*)alloc((size_t)DIMn * HIDn * 2);
  h16*   Ct    = (h16*)alloc((size_t)DIMn * HIDn * 2);
  h16*   Gt    = (h16*)alloc((size_t)DIMn * HIDn * 2);
  float* delta = (float*)alloc((size_t)BS * 4);
  float* segA  = (float*)alloc((size_t)NSEG * BATCH * HIDn * 4);
  float* segS  = (float*)alloc((size_t)NSEG * BATCH * HIDn * 4);
  float* segSt = (float*)alloc((size_t)NSEG * BATCH * HIDn * 4);

  const dim3 t256(256);
  convertT_kernel<<<dim3(64, 64), t256, 0, stream>>>(Bw, Bt);
  convertT_kernel<<<dim3(64, 64), t256, 0, stream>>>(Cw, Ct);
  convertT_kernel<<<dim3(64, 64), t256, 0, stream>>>(Gw, Gt);

  ln_delta_kernel<<<dim3(BS), t256, 0, stream>>>(x, lns, lnb, dtk, dtb, xn16, delta);

  // U = xn @ B  (f16 out)
  wmma_gemm_kernel<0><<<dim3(16, 256), t256, 0, stream>>>(xn16, Bt, nullptr, U16, nullptr);

  // segmented diagonal-A scan
  scan_seg_kernel<<<dim3(8, 8, NSEG), t256, 0, stream>>>(U16, delta, Av, segA, segS);
  scan_combine_kernel<<<dim3(8, 8), t256, 0, stream>>>(segA, segS, ctx, segSt,
                                                       out + (size_t)BS * DIMn);
  scan_emit_kernel<<<dim3(8, 8, NSEG), t256, 0, stream>>>(U16, delta, Av, segSt, s16);

  // outs = states @ C  (f32 to d_out + f16 copy)
  wmma_gemm_kernel<1><<<dim3(16, 256), t256, 0, stream>>>(s16, Ct, out, o16, nullptr);

  // gate = outs @ G + gb ; d_out = outs * sigmoid(gate)  (in-place epilogue)
  wmma_gemm_kernel<2><<<dim3(16, 256), t256, 0, stream>>>(o16, Gt, out, nullptr, gb);
}
